// AdaPT_Linear_36060545417721
// MI455X (gfx1250) — 8-bit matrix operand order repaired and checked by reading, compiled, not run on hardware
//
#include <hip/hip_runtime.h>

typedef __attribute__((ext_vector_type(8))) int v8i;
typedef __attribute__((ext_vector_type(4))) int v4i;
typedef __attribute__((ext_vector_type(2))) int v2i;

#define QL 127.0f
#define BM 128
#define BN 128
#define BK 64
#define LDSS 80

__device__ __forceinline__ void async_copy_b128(unsigned ldsOff,
                                                const signed char* g) {
  asm volatile("global_load_async_to_lds_b128 %0, %1, off"
               :
               : "v"(ldsOff), "v"(g)
               : "memory");
}
__device__ __forceinline__ void wait_async0() {
  asm volatile("s_wait_asynccnt 0" ::: "memory");
}
__device__ __forceinline__ unsigned lds_addr(const void* p) {
  return (unsigned)(unsigned long long)p;
}

__global__ __launch_bounds__(256) void quant16_kernel(
    const float* __restrict__ in, signed char* __restrict__ q,
    const float* __restrict__ amax, long long ngroups) {
  long long i = (long long)blockIdx.x * blockDim.x + threadIdx.x;
  if (i >= ngroups) return;
  const float s = QL / amax[0];
  const float4* p = (const float4*)in + i * 4;
  union { signed char c[16]; int4 v; } u;
#pragma unroll
  for (int g = 0; g < 4; ++g) {
    float4 f = p[g];
    float vals[4] = {f.x, f.y, f.z, f.w};
#pragma unroll
    for (int e = 0; e < 4; ++e) {
      float t = fminf(fmaxf(vals[e] * s, -QL), QL);
      u.c[g * 4 + e] = (signed char)(int)rintf(t);
    }
  }
  ((int4*)q)[i] = u.v;
}

__global__ __launch_bounds__(256) void i8gemm_wmma_kernel(
    const signed char* __restrict__ qx, const signed char* __restrict__ qw,
    const float* __restrict__ bias, const float* __restrict__ amax,
    const float* __restrict__ amax_w, float* __restrict__ out,
    int M, int N, int K) {
  __shared__ signed char sA[2][BM * LDSS];
  __shared__ signed char sB[2][BN * LDSS];

  const int t = threadIdx.x;
  const int lane = t & 31;
  const int wave = t >> 5;
  const int waveM = wave >> 1;
  const int waveN = wave & 1;
  const int lrow = lane & 15;
  const int lhi = lane >> 4;

  const int mBase = blockIdx.y * BM;
  const int nBase = blockIdx.x * BN;

  v8i acc[2][4];
  const v8i vzero = {};
#pragma unroll
  for (int i = 0; i < 2; ++i)
#pragma unroll
    for (int j = 0; j < 4; ++j) acc[i][j] = vzero;

  const int ldRow = t >> 2;
  const int ldChunk = (t & 3) * 16;
  const signed char* gA0 = qx + (size_t)(mBase + ldRow) * K + ldChunk;
  const signed char* gA1 = gA0 + (size_t)64 * K;
  const signed char* gB0 = qw + (size_t)(nBase + ldRow) * K + ldChunk;
  const signed char* gB1 = gB0 + (size_t)64 * K;
  const int sOff0 = ldRow * LDSS + ldChunk;
  const int sOff1 = (64 + ldRow) * LDSS + ldChunk;

  auto stage = [&](int b, int k) {
    async_copy_b128(lds_addr(&sA[b][sOff0]), gA0 + k);
    async_copy_b128(lds_addr(&sA[b][sOff1]), gA1 + k);
    async_copy_b128(lds_addr(&sB[b][sOff0]), gB0 + k);
    async_copy_b128(lds_addr(&sB[b][sOff1]), gB1 + k);
  };

  auto compute = [&](int b) {
    v8i afrag[2];
#pragma unroll
    for (int i = 0; i < 2; ++i) {
      const signed char* ap =
          &sA[b][(waveM * 32 + i * 16 + lrow) * LDSS + lhi * 16];
      v2i a0 = *(const v2i*)(ap + 0);
      v2i a1 = *(const v2i*)(ap + 8);
      v2i a2 = *(const v2i*)(ap + 32);
      v2i a3 = *(const v2i*)(ap + 40);
      v8i a;
      a[0] = a0.x; a[1] = a0.y; a[2] = a1.x; a[3] = a1.y;
      a[4] = a2.x; a[5] = a2.y; a[6] = a3.x; a[7] = a3.y;
      afrag[i] = a;
    }
    v8i bfrag[4];
#pragma unroll
    for (int j = 0; j < 4; ++j) {
      const signed char* bp =
          &sB[b][(waveN * 64 + j * 16 + lrow) * LDSS + lhi * 16];
      v4i b0 = *(const v4i*)(bp + 0);
      v4i b1 = *(const v4i*)(bp + 32);
      v8i bb;
      bb[0] = b0.x; bb[1] = b0.y; bb[2] = b0.z; bb[3] = b0.w;
      bb[4] = b1.x; bb[5] = b1.y; bb[6] = b1.z; bb[7] = b1.w;
      bfrag[j] = bb;
    }
#pragma unroll
    for (int j = 0; j < 4; ++j)
#pragma unroll
      for (int i = 0; i < 2; ++i)
        acc[i][j] = __builtin_amdgcn_wmma_i32_16x16x64_iu8(
            true, afrag[i], true, bfrag[j], acc[i][j], false, false);
  };

  stage(0, 0);
  wait_async0();
  __syncthreads();

  for (int k0 = 0; k0 < K; k0 += 2 * BK) {
    stage(1, k0 + BK);
    compute(0);
    wait_async0();
    __syncthreads();

    const bool more = (k0 + 2 * BK) < K;
    if (more) stage(0, k0 + 2 * BK);
    compute(1);
    if (more) {
      wait_async0();
      __syncthreads();
    }
  }

  const float inv_scale = (amax[0] * amax_w[0]) / (QL * QL);
#pragma unroll
  for (int i = 0; i < 2; ++i) {
    const int m0 = mBase + waveM * 32 + i * 16 + lhi * 8;
#pragma unroll
    for (int j = 0; j < 4; ++j) {
      const int n = nBase + waveN * 64 + j * 16 + lrow;
      const float bn = bias[n];
      float* op = out + (size_t)m0 * N + n;
#pragma unroll
      for (int r = 0; r < 8; ++r) {
        op[(size_t)r * N] = (float)acc[i][j][r] * inv_scale + bn;
      }
    }
  }
}

extern "C" void kernel_launch(void* const* d_in, const int* in_sizes, int n_in,
                              void* d_out, int out_size, void* d_ws,
                              size_t ws_size, hipStream_t stream) {
  (void)n_in; (void)out_size; (void)ws_size;
  const float* x      = (const float*)d_in[0];
  const float* w      = (const float*)d_in[1];
  const float* bias   = (const float*)d_in[2];
  const float* amax   = (const float*)d_in[3];
  const float* amax_w = (const float*)d_in[4];
  float* out = (float*)d_out;

  const long long xElems = in_sizes[0];
  const long long wElems = in_sizes[1];
  const int N = in_sizes[2];
  const int K = (int)(wElems / N);
  const int M = (int)(xElems / K);

  signed char* qx = (signed char*)d_ws;
  signed char* qw = qx + (size_t)xElems;

  const long long xg = xElems / 16;
  const long long wg = wElems / 16;
  quant16_kernel<<<(unsigned)((xg + 255) / 256), 256, 0, stream>>>(x, qx, amax, xg);
  quant16_kernel<<<(unsigned)((wg + 255) / 256), 256, 0, stream>>>(w, qw, amax_w, wg);

  dim3 grid(N / BN, M / BM);
  i8gemm_wmma_kernel<<<grid, 256, 0, stream>>>(qx, qw, bias, amax, amax_w, out,
                                               M, N, K);
}
